// MultiHeadAttentionLayerGritSparse_4114578669930
// MI455X (gfx1250) — matrix operand order repaired and checked by reading, not compiled, not run on hardware
//
#include <hip/hip_runtime.h>
#include <hip/hip_bf16.h>
#include <cstdint>
#include <cstddef>

// ---------------------------------------------------------------------------
// GRIT sparse multi-head attention layer for MI455X (gfx1250, wave32, WMMA).
// Dense GEMMs -> v_wmma_f32_16x16x32_bf16 (f32 accumulate, fused epilogues).
// Sparse gather / segmented softmax / scatter-add -> f32 VALU, L2-resident.
// ---------------------------------------------------------------------------

typedef __bf16 bf16_t;
typedef __attribute__((ext_vector_type(16))) __bf16 v16bf;
typedef __attribute__((ext_vector_type(8)))  float  v8f;

#define LN_EPS  1e-5f
#define CLAMP_V 5.0f

// order-preserving f32 <-> u32 mapping for atomic segment-max
__device__ __forceinline__ unsigned enc_ord(float f) {
  int i = __float_as_int(f);
  return (i >= 0) ? ((unsigned)i | 0x80000000u) : ~((unsigned)i);
}
__device__ __forceinline__ float dec_ord(unsigned u) {
  int i = (u & 0x80000000u) ? (int)(u & 0x7FFFFFFFu) : (int)(~u);
  return __int_as_float(i);
}

// ---------------------------------------------------------------------------
// Generic WMMA GEMM:  C[M,NOUT] = epilogue( A[M,K] @ W[K,NOUT] )
// Weights stored [in,out] (row-major K x NOUT), matching the reference.
// Each wave owns one 16-col tile (weight fragments pinned in registers) and
// grid-strides over 16-row tiles. M, NOUT, K all multiples of 16/32 here.
// Epilogue (bias / residual-add / ReLU) selected at compile time -> the hot
// loop is branch-free straight-line code.
// ---------------------------------------------------------------------------
template <int K, int NOUT, bool BIAS, bool RESID, bool RELU>
__global__ __launch_bounds__(256)
void gemm_wmma_bf16(const float* __restrict__ A, const float* __restrict__ W,
                    const float* __restrict__ bias, const float* __restrict__ resid,
                    float* __restrict__ C, int M)
{
  constexpr int KSTEPS = K / 32;
  constexpr int NTILES = NOUT / 16;

  const int lane       = threadIdx.x & 31;
  const int wave       = blockIdx.x * (blockDim.x >> 5) + (threadIdx.x >> 5);
  const int totalWaves = gridDim.x * (blockDim.x >> 5);
  const int Mtiles     = M >> 4;
  const int nt         = wave % NTILES;           // N tile owned by this wave
  const int mt0        = wave / NTILES;
  const int mStride    = totalWaves / NTILES;     // totalWaves % NTILES == 0 by construction
  const int g          = lane >> 4;               // lane group (0/1)
  const int nn         = lane & 15;
  const int col        = nt * 16 + nn;

  // B fragments (weights): element e <-> K = step*32 + g*16 + e, lane -> column.
  v16bf bfrag[KSTEPS];
#pragma unroll
  for (int s = 0; s < KSTEPS; ++s) {
    const float* wp = W + (size_t)(s * 32 + g * 16) * NOUT + col;
#pragma unroll
    for (int j = 0; j < 8; ++j) {
      bfrag[s][2 * j]     = (bf16_t)wp[(2 * j) * NOUT];
      bfrag[s][2 * j + 1] = (bf16_t)wp[(2 * j + 1) * NOUT];
    }
  }
  const float biasv = BIAS ? bias[col] : 0.0f;

  for (int mt = mt0; mt < Mtiles; mt += mStride) {
    const float* arow = A + (size_t)(mt * 16 + nn) * K;   // A: lane -> row M = lane%16
    v8f acc = {0.f, 0.f, 0.f, 0.f, 0.f, 0.f, 0.f, 0.f};
#pragma unroll
    for (int s = 0; s < KSTEPS; ++s) {
      v16bf afrag;
#pragma unroll
      for (int j = 0; j < 8; ++j) {
        // A layout: pair j -> K = 16*g + 2j; pairs are K-adjacent.
        const int   k0 = s * 32 + 16 * g + 2 * j;
        const float2 v = *(const float2*)(arow + k0);
        afrag[2 * j]     = (bf16_t)v.x;
        afrag[2 * j + 1] = (bf16_t)v.y;
      }
      acc = __builtin_amdgcn_wmma_f32_16x16x32_bf16(
          false, afrag, false, bfrag[s], (short)0, acc, false, false);
    }
    // C layout: VGPR r, lane -> row = mt*16 + r + 8*g, col = nt*16 + nn
#pragma unroll
    for (int r = 0; r < 8; ++r) {
      const int orow = mt * 16 + r + 8 * g;
      float v = acc[r] + biasv;
      if (RESID) v += resid[(size_t)orow * NOUT + col];
      if (RELU)  v = fmaxf(v, 0.f);
      C[(size_t)orow * NOUT + col] = v;
    }
  }
}

// ---------------------------------------------------------------------------
// LayerNorm over rows of 64 (one wave per row, 2 elems/lane, shfl reductions)
// ---------------------------------------------------------------------------
__global__ __launch_bounds__(256)
void layernorm64_kernel(float* __restrict__ X, const float* __restrict__ gamma,
                        const float* __restrict__ beta, int M)
{
  const int wave = blockIdx.x * (blockDim.x >> 5) + (threadIdx.x >> 5);
  const int lane = threadIdx.x & 31;
  if (wave >= M) return;
  float* row = X + (size_t)wave * 64;
  float2 v = *(const float2*)(row + lane * 2);
  float sum = v.x + v.y;
#pragma unroll
  for (int o = 16; o > 0; o >>= 1) sum += __shfl_xor(sum, o, 32);
  const float mean = sum * (1.0f / 64.0f);
  const float dx = v.x - mean, dy = v.y - mean;
  float vs = dx * dx + dy * dy;
#pragma unroll
  for (int o = 16; o > 0; o >>= 1) vs += __shfl_xor(vs, o, 32);
  const float rstd = rsqrtf(vs * (1.0f / 64.0f) + LN_EPS);
  const float2 gg = *(const float2*)(gamma + lane * 2);
  const float2 bb = *(const float2*)(beta + lane * 2);
  float2 outv;
  outv.x = dx * rstd * gg.x + bb.x;
  outv.y = dy * rstd * gg.y + bb.y;
  *(float2*)(row + lane * 2) = outv;
}

// ---------------------------------------------------------------------------
// init: segment-max keys, denominators, aggregation buffer
// ---------------------------------------------------------------------------
__global__ void init_kernel(unsigned* __restrict__ smax_u, float* __restrict__ denom,
                            float* __restrict__ x_agg, int NH, int ND)
{
  const int i = blockIdx.x * blockDim.x + threadIdx.x;
  if (i < NH) { smax_u[i] = 0u; denom[i] = 0.f; }   // enc_ord key 0 == -inf sentinel
  if (i < ND) x_agg[i] = 0.f;
}

// ---------------------------------------------------------------------------
// Pass A: per (edge, head): score modulation, signed-sqrt, relu, e_out,
//         per-head scalar s (clip) + atomic segment-max.
// ---------------------------------------------------------------------------
__global__ __launch_bounds__(256)
void edge_score_kernel(const float* __restrict__ xk, const float* __restrict__ xq,
                       const float* __restrict__ eb, const float* __restrict__ ew,
                       const int* __restrict__ ei, const float* __restrict__ Aw,
                       float* __restrict__ e_out, float* __restrict__ s_buf,
                       unsigned* __restrict__ smax_u, int E)
{
  const int tid = blockIdx.x * blockDim.x + threadIdx.x;
  if (tid >= E * 8) return;
  const int e = tid >> 3, h = tid & 7;
  const int src = ei[e], dst = ei[E + e];
  const size_t eoff = (size_t)e * 64 + h * 8;

  float k[8], q[8], w[8], b[8], o[8];
  *(float4*)&k[0] = *(const float4*)(xk + (size_t)src * 64 + h * 8);
  *(float4*)&k[4] = *(const float4*)(xk + (size_t)src * 64 + h * 8 + 4);
  *(float4*)&q[0] = *(const float4*)(xq + (size_t)dst * 64 + h * 8);
  *(float4*)&q[4] = *(const float4*)(xq + (size_t)dst * 64 + h * 8 + 4);
  *(float4*)&w[0] = *(const float4*)(ew + eoff);
  *(float4*)&w[4] = *(const float4*)(ew + eoff + 4);
  *(float4*)&b[0] = *(const float4*)(eb + eoff);
  *(float4*)&b[4] = *(const float4*)(eb + eoff + 4);

  float sacc = 0.f;
#pragma unroll
  for (int d = 0; d < 8; ++d) {
    float sc = (k[d] + q[d]) * w[d];
    sc = copysignf(sqrtf(fabsf(sc)), sc);   // sign(s)*sqrt(|s|)
    const float v = fmaxf(sc + b[d], 0.f);
    o[d] = v;
    sacc += v * Aw[d * 8 + h];              // einsum with Aw[d][h]
  }
  *(float4*)(e_out + eoff)     = *(const float4*)&o[0];
  *(float4*)(e_out + eoff + 4) = *(const float4*)&o[4];

  const float s = fminf(fmaxf(sacc, -CLAMP_V), CLAMP_V);
  s_buf[tid] = s;
  atomicMax(smax_u + (size_t)dst * 8 + h, enc_ord(s));
}

// Pass B: w = exp(s - max), atomic denominator
__global__ __launch_bounds__(256)
void edge_exp_kernel(float* __restrict__ s_buf, const unsigned* __restrict__ smax_u,
                     float* __restrict__ denom, const int* __restrict__ ei, int E)
{
  const int tid = blockIdx.x * blockDim.x + threadIdx.x;
  if (tid >= E * 8) return;
  const int e = tid >> 3, h = tid & 7;
  const int dst = ei[E + e];
  const float m = dec_ord(smax_u[(size_t)dst * 8 + h]);
  const float w = __expf(s_buf[tid] - m);
  s_buf[tid] = w;
  atomicAdd(denom + (size_t)dst * 8 + h, w);
}

// Pass C: normalized weighted scatter-add of xv into x_agg
__global__ __launch_bounds__(256)
void edge_agg_kernel(const float* __restrict__ s_buf, const float* __restrict__ denom,
                     const float* __restrict__ xv, const int* __restrict__ ei,
                     float* __restrict__ x_agg, int E)
{
  const int tid = blockIdx.x * blockDim.x + threadIdx.x;
  if (tid >= E * 8) return;
  const int e = tid >> 3, h = tid & 7;
  const int src = ei[e], dst = ei[E + e];
  const float dn = denom[(size_t)dst * 8 + h] + 1e-16f;
  const float sc = s_buf[tid] / dn;
  const float* vp = xv + (size_t)src * 64 + h * 8;
  float* ap = x_agg + (size_t)dst * 64 + h * 8;
#pragma unroll
  for (int d = 0; d < 8; ++d) atomicAdd(ap + d, sc * vp[d]);
}

// ---------------------------------------------------------------------------
// Host-side orchestration
// ---------------------------------------------------------------------------
static inline int gemm_blocks(int M, int nout) {
  long tiles  = (long)(M / 16) * (nout / 16);
  long blocks = (tiles + 7) / 8;               // 8 waves per 256-thread block
  if (blocks > 4096) blocks = 4096;
  if (blocks < 1)    blocks = 1;
  return (int)blocks;
}

extern "C" void kernel_launch(void* const* d_in, const int* in_sizes, int n_in,
                              void* d_out, int out_size, void* d_ws, size_t ws_size,
                              hipStream_t stream)
{
  (void)n_in; (void)out_size; (void)ws_size;

  const float* x      = (const float*)d_in[0];
  const int*   ei     = (const int*)  d_in[1];
  const float* npe    = (const float*)d_in[2];
  const float* wq_w   = (const float*)d_in[3];
  const float* wq_b   = (const float*)d_in[4];
  const float* wk_w   = (const float*)d_in[5];
  const float* wv_w   = (const float*)d_in[6];
  const float* web_w  = (const float*)d_in[7];
  const float* web_b  = (const float*)d_in[8];
  const float* wew_w  = (const float*)d_in[9];
  const float* wew_b  = (const float*)d_in[10];
  const float* wo_w   = (const float*)d_in[11];
  const float* weo_w  = (const float*)d_in[12];
  const float* Aw     = (const float*)d_in[13];
  const float* nf_w1  = (const float*)d_in[14];
  const float* nf_b1  = (const float*)d_in[15];
  const float* nf_w2  = (const float*)d_in[16];
  const float* nf_b2  = (const float*)d_in[17];
  const float* ef_w1  = (const float*)d_in[18];
  const float* ef_b1  = (const float*)d_in[19];
  const float* ef_w2  = (const float*)d_in[20];
  const float* ef_b2  = (const float*)d_in[21];
  const float* nattn_g= (const float*)d_in[22];
  const float* nattn_b= (const float*)d_in[23];
  const float* nffn_g = (const float*)d_in[24];
  const float* nffn_b = (const float*)d_in[25];
  const float* eattn_g= (const float*)d_in[26];
  const float* eattn_b= (const float*)d_in[27];
  const float* effn_g = (const float*)d_in[28];
  const float* effn_b = (const float*)d_in[29];

  const int N = in_sizes[0] / 64;      // 50000 (multiple of 16)
  const int E = in_sizes[1] / 2;       // 800000 (multiple of 16)

  // ---- workspace carve-out (f32 elements) ----
  float* ws = (float*)d_ws;
  size_t o = 0;
  float* xq    = ws + o; o += (size_t)N * 64;
  float* xk    = ws + o; o += (size_t)N * 64;
  float* xv    = ws + o; o += (size_t)N * 64;
  float* ew    = ws + o; o += (size_t)E * 64;     // ew + e_out are adjacent:
  float* e_out = ws + o; o += (size_t)E * 64;     //   reused later as E x 128 hidden
  float* eb    = ws + o; o += (size_t)E * 64;
  float* s_buf = ws + o; o += (size_t)E * 8;
  unsigned* smax_u = (unsigned*)(ws + o); o += (size_t)N * 8;
  float* denom = ws + o; o += (size_t)N * 8;
  float* x_agg = ws + o; o += (size_t)N * 64;
  float* h_buf = ws + o; o += (size_t)N * 64;
  float* nhid  = xq;                               // N x 128 (reuses xq+xk)
  float* ehid  = ew;                               // E x 128 (reuses ew+e_out)

  float* out_h = (float*)d_out;                    // [N,64]
  float* out_e = (float*)d_out + (size_t)N * 64;   // [E,64]

  const int TB = 256;
  const int ehBlocks = (E * 8 + TB - 1) / TB;

  // init segment buffers + aggregation buffer
  init_kernel<<<(N * 64 + TB - 1) / TB, TB, 0, stream>>>(smax_u, denom, x_agg, N * 8, N * 64);

  // node projections: xq = x@wq + bq ; xk = x@wk ; xv = x@wv
  gemm_wmma_bf16<64, 64, true,  false, false><<<gemm_blocks(N, 64), TB, 0, stream>>>(x, wq_w, wq_b, nullptr, xq, N);
  gemm_wmma_bf16<64, 64, false, false, false><<<gemm_blocks(N, 64), TB, 0, stream>>>(x, wk_w, nullptr, nullptr, xk, N);
  gemm_wmma_bf16<64, 64, false, false, false><<<gemm_blocks(N, 64), TB, 0, stream>>>(x, wv_w, nullptr, nullptr, xv, N);

  // edge projections: eb = npe@web + b ; ew = npe@wew + b
  gemm_wmma_bf16<64, 64, true,  false, false><<<gemm_blocks(E, 64), TB, 0, stream>>>(npe, web_w, web_b, nullptr, eb, E);
  gemm_wmma_bf16<64, 64, true,  false, false><<<gemm_blocks(E, 64), TB, 0, stream>>>(npe, wew_w, wew_b, nullptr, ew, E);

  // sparse attention core
  edge_score_kernel<<<ehBlocks, TB, 0, stream>>>(xk, xq, eb, ew, ei, Aw, e_out, s_buf, smax_u, E);
  edge_exp_kernel  <<<ehBlocks, TB, 0, stream>>>(s_buf, smax_u, denom, ei, E);
  edge_agg_kernel  <<<ehBlocks, TB, 0, stream>>>(s_buf, denom, xv, ei, x_agg, E);

  // edge output projection with residual: e = e_out@weo + npe  (overwrites eb)
  gemm_wmma_bf16<64, 64, false, true,  false><<<gemm_blocks(E, 64), TB, 0, stream>>>(e_out, weo_w, nullptr, npe, eb, E);
  // node output projection with residual: h = x_agg@wo + x
  gemm_wmma_bf16<64, 64, false, true,  false><<<gemm_blocks(N, 64), TB, 0, stream>>>(x_agg, wo_w, nullptr, x, h_buf, N);

  // attention LayerNorms (in place)
  layernorm64_kernel<<<(N + 7) / 8, TB, 0, stream>>>(h_buf, nattn_g, nattn_b, N);
  layernorm64_kernel<<<(E + 7) / 8, TB, 0, stream>>>(eb, eattn_g, eattn_b, E);

  // node FFN: nhid = relu(h@nf_w1+b1) ; h = nhid@nf_w2 + b2 + h ; LN
  gemm_wmma_bf16<64, 128, true, false, true ><<<gemm_blocks(N, 128), TB, 0, stream>>>(h_buf, nf_w1, nf_b1, nullptr, nhid, N);
  gemm_wmma_bf16<128, 64, true, true,  false><<<gemm_blocks(N, 64),  TB, 0, stream>>>(nhid, nf_w2, nf_b2, h_buf, out_h, N);
  layernorm64_kernel<<<(N + 7) / 8, TB, 0, stream>>>(out_h, nffn_g, nffn_b, N);

  // edge FFN: ehid = relu(e@ef_w1+b1) ; e = ehid@ef_w2 + b2 + e ; LN
  gemm_wmma_bf16<64, 128, true, false, true ><<<gemm_blocks(E, 128), TB, 0, stream>>>(eb, ef_w1, ef_b1, nullptr, ehid, E);
  gemm_wmma_bf16<128, 64, true, true,  false><<<gemm_blocks(E, 64),  TB, 0, stream>>>(ehid, ef_w2, ef_b2, eb, out_e, E);
  layernorm64_kernel<<<(E + 7) / 8, TB, 0, stream>>>(out_e, effn_g, effn_b, E);
}
